// SDEActor_90864328114958
// MI455X (gfx1250) — matrix operand order repaired and checked by reading, not compiled, not run on hardware
//
#include <hip/hip_runtime.h>
#include <hip/hip_bf16.h>

typedef __attribute__((ext_vector_type(16))) _Float16 v16h;
typedef __attribute__((ext_vector_type(8)))  _Float16 v8h;
typedef __attribute__((ext_vector_type(8)))  float    v8f;

#define LANE (threadIdx.x & 31)
#define WAVE ((int)(threadIdx.x >> 5))

__device__ __forceinline__ v8f wmma_f16(v16h a, v16h b, v8f c) {
  return __builtin_amdgcn_wmma_f32_16x16x32_f16(false, a, false, b, (short)0, c, false, false);
}

// Within-tile swizzle offsets, contiguous k order: lane half h holds k = 16h + idx.
// A tile 16x32 (m4 = row 0..15, k5 = k 0..31 within 32-block):
__device__ __forceinline__ int swzA(int m4, int k5) {
  int lane = (m4 & 15) + (((k5 >> 4) & 1) << 4);
  int idx  = (k5 & 15);
  return lane * 16 + idx;
}
// B tile 32x16 (k5 = k 0..31 within block, n4 = col 0..15):
__device__ __forceinline__ int swzB(int k5, int n4) {
  int lane = (n4 & 15) + (((k5 >> 4) & 1) << 4);
  return lane * 16 + (k5 & 15);
}
// Contiguous per-lane tile load (32B -> 2x ds_load_b128)
__device__ __forceinline__ v16h ldtile(const _Float16* base) {
  return *(const v16h*)(base + LANE * 16);
}

// ---------------------------------------------------------------------------
// Kernel 1: fused conv1->conv2->conv3 (BN-eval+ReLU) -> mean pool.
// One block per batch element. Activations stored transposed [pos][chan] so
// conv B-tiles (k = tap*C + chan ordering) are contiguous per-lane v16h loads;
// weights pre-swizzled into WMMA A register order.
// ---------------------------------------------------------------------------
__global__ __launch_bounds__(256) void k_conv(
    const float* __restrict__ state,
    const float* __restrict__ c1w, const float* __restrict__ c1b,
    const float* __restrict__ bn1g, const float* __restrict__ bn1b,
    const float* __restrict__ c2w, const float* __restrict__ c2b,
    const float* __restrict__ bn2g, const float* __restrict__ bn2b,
    const float* __restrict__ c3w, const float* __restrict__ c3b,
    const float* __restrict__ bn3g, const float* __restrict__ bn3b,
    _Float16* __restrict__ Xout) {
  __shared__ alignas(32) _Float16 sW2s[12 * 512];   // 4 mt x 3 kk tiles
  __shared__ alignas(32) _Float16 sW3s[24 * 512];   // 4 mt x 6 kk tiles
  __shared__ alignas(32) _Float16 act1T[66 * 32];   // [pos][chan], rows 0,51.. zero
  __shared__ alignas(32) _Float16 act2T[66 * 64];
  __shared__ float sw1[96];
  __shared__ float s1[32], t1[32];
  __shared__ float s2[64], t2[64], s3[64], t3[64];
  __shared__ float pooled[30 * 64];
  __shared__ float inrow[52];

  const int b = blockIdx.x, tid = threadIdx.x;
  // conv2 weights: k = t*32 + c  ->  source c2w[m*96 + c*3 + t], swizzled A tiles
  for (int i = tid; i < 64 * 96; i += 256) {
    int m = i / 96, k = i - m * 96;
    int t = k >> 5, c = k & 31;
    sW2s[((m >> 4) * 3 + (k >> 5)) * 512 + swzA(m & 15, k & 31)] =
        (_Float16)c2w[m * 96 + c * 3 + t];
  }
  // conv3 weights: k = t*64 + c
  for (int i = tid; i < 64 * 192; i += 256) {
    int m = i / 192, k = i - m * 192;
    int t = k >> 6, c = k & 63;
    sW3s[((m >> 4) * 6 + (k >> 5)) * 512 + swzA(m & 15, k & 31)] =
        (_Float16)c3w[m * 192 + c * 3 + t];
  }
  for (int i = tid; i < 96; i += 256) sw1[i] = c1w[i];
  if (tid < 32) {
    float s = bn1g[tid] * rsqrtf(1.f + 1e-5f);
    s1[tid] = s; t1[tid] = s * c1b[tid] + bn1b[tid];
  }
  if (tid < 64) {
    float sa = bn2g[tid] * rsqrtf(1.f + 1e-5f);
    s2[tid] = sa; t2[tid] = sa * c2b[tid] + bn2b[tid];
    float sb = bn3g[tid] * rsqrtf(1.f + 1e-5f);
    s3[tid] = sb; t3[tid] = sb * c3b[tid] + bn3b[tid];
  }
  for (int i = tid; i < 66 * 32; i += 256) act1T[i] = (_Float16)0.f;
  for (int i = tid; i < 66 * 64; i += 256) act2T[i] = (_Float16)0.f;
  for (int i = tid; i < 30 * 64; i += 256) pooled[i] = 0.f;
  __syncthreads();

  const float* prow = state + (size_t)b * 1531;
  for (int a = 0; a < 30; ++a) {
    if (tid < 50) inrow[tid] = prow[a * 50 + tid];
    __syncthreads();
    // conv1 (scalar) + BN + ReLU -> act1T[pos+1][c]
    for (int idx = tid; idx < 32 * 50; idx += 256) {
      int c = idx >> 5 ? 0 : 0;  // placeholder to keep compiler honest
      c = idx / 50; int x = idx - c * 50;
      float acc = sw1[c * 3 + 1] * inrow[x];
      if (x > 0)  acc += sw1[c * 3 + 0] * inrow[x - 1];
      if (x < 49) acc += sw1[c * 3 + 2] * inrow[x + 1];
      act1T[(x + 1) * 32 + c] = (_Float16)fmaxf(s1[c] * acc + t1[c], 0.f);
    }
    __syncthreads();
    // conv2: C(64x64) = W2(64x96) x B(96x64); B tile = 16 consecutive chans
    for (int tile = WAVE; tile < 16; tile += 8) {
      int mt = tile & 3, nt = tile >> 2;
      int n = nt * 16 + (LANE & 15);
      v8f acc = {};
#pragma unroll
      for (int kk = 0; kk < 3; ++kk) {  // kk == tap
        v16h Am = ldtile(sW2s + (mt * 3 + kk) * 512);
        v16h Bm = *(const v16h*)(act1T + (n + kk) * 32 + (LANE & 16));
        acc = wmma_f16(Am, Bm, acc);
      }
      int mb = mt * 16 + ((LANE & 16) ? 8 : 0);
      if (n < 50) {
        v8h o;
#pragma unroll
        for (int r = 0; r < 8; ++r) {
          int m = mb + r;
          o[r] = (_Float16)fmaxf(s2[m] * acc[r] + t2[m], 0.f);
        }
        *(v8h*)(act2T + (n + 1) * 64 + mb) = o;  // 16B packed store
      }
    }
    __syncthreads();
    // conv3: K=192, k = tap*64 + c
    for (int tile = WAVE; tile < 16; tile += 8) {
      int mt = tile & 3, nt = tile >> 2;
      int n = nt * 16 + (LANE & 15);
      v8f acc = {};
#pragma unroll
      for (int kk = 0; kk < 6; ++kk) {
        int t = kk >> 1;
        int cb = ((kk & 1) << 5) + (LANE & 16);
        v16h Am = ldtile(sW3s + (mt * 6 + kk) * 512);
        v16h Bm = *(const v16h*)(act2T + (n + t) * 64 + cb);
        acc = wmma_f16(Am, Bm, acc);
      }
      int mb = mt * 16 + ((LANE & 16) ? 8 : 0);
      if (n < 50) {
#pragma unroll
        for (int r = 0; r < 8; ++r) {
          int m = mb + r;
          float y = fmaxf(s3[m] * acc[r] + t3[m], 0.f);
          atomicAdd(&pooled[a * 64 + m], y * 0.02f);  // /50
        }
      }
    }
    __syncthreads();
  }
  // flatten order of x.reshape(B,-1) is [ch][asset]
  for (int i = tid; i < 1920; i += 256) {
    int m = i / 30, a2 = i - m * 30;
    Xout[(size_t)b * 1920 + i] = (_Float16)pooled[a2 * 64 + m];
  }
}

// ---------------------------------------------------------------------------
// Kernel 2: fc(1920->128) -> concat wts -> init MLP (159->128->64->31) with
// WMMA; activations live in LDS in swizzled A-tile layout, B panels staged
// swizzled per K-step. 64 batch rows/block; scalar diffusion head -> sigma.
// ---------------------------------------------------------------------------
__global__ __launch_bounds__(256) void k_head(
    const _Float16* __restrict__ Xg, const float* __restrict__ state,
    const float* __restrict__ fcw, const float* __restrict__ fcb,
    const float* __restrict__ iw1, const float* __restrict__ ib1,
    const float* __restrict__ iw2, const float* __restrict__ ib2,
    const float* __restrict__ iw3, const float* __restrict__ ib3,
    const float* __restrict__ dfw1, const float* __restrict__ dfb1,
    const float* __restrict__ dflg, const float* __restrict__ dflb,
    const float* __restrict__ dfw2, const float* __restrict__ dfb2,
    float* __restrict__ traj, float* __restrict__ sigma_ws) {
  __shared__ alignas(32) _Float16 sBs[8 * 512];    // up to 8 N-tiles of B
  __shared__ alignas(32) _Float16 X0s[20 * 512];   // 4 mt x 5 kk (width 160)
  __shared__ alignas(32) _Float16 X1s[16 * 512];   // 4 mt x 4 kk (width 128)
  __shared__ float a0s[64 * 32];

  const int tid = threadIdx.x;
  const int r0 = blockIdx.x * 64;
  const int mt = WAVE & 3;

  // ---- fc: (64x1920) @ (1920x128); A tiles straight from global (2x b128) --
  {
    v8f acc[4] = {};
    const _Float16* Arow =
        Xg + (size_t)(r0 + mt * 16 + (LANE & 15)) * 1920 + ((LANE & 16) ? 16 : 0);
    for (int kk = 0; kk < 60; ++kk) {
      for (int i = tid; i < 32 * 128; i += 256) {
        int kr = i >> 7, n = i & 127;
        sBs[(n >> 4) * 512 + swzB(kr, n & 15)] =
            (_Float16)fcw[(size_t)(kk * 32 + kr) * 128 + n];
      }
      if (kk < 59 && tid < 32)
        __builtin_prefetch(fcw + (size_t)((kk + 1) * 32 + tid) * 128, 0, 1);
      __syncthreads();
      union { v16h v; v8h h[2]; } ua;
      ua.h[0] = *(const v8h*)(Arow + kk * 32);
      ua.h[1] = *(const v8h*)(Arow + kk * 32 + 8);
#pragma unroll
      for (int t = 0; t < 4; ++t) {
        int nt = (WAVE >> 2) + 2 * t;
        acc[t] = wmma_f16(ua.v, ldtile(sBs + nt * 512), acc[t]);
      }
      __syncthreads();
    }
#pragma unroll
    for (int t = 0; t < 4; ++t) {
      int nt = (WAVE >> 2) + 2 * t;
      int n = nt * 16 + (LANE & 15), mb = (LANE & 16) ? 8 : 0;
      float bias = fcb[n];
#pragma unroll
      for (int r = 0; r < 8; ++r)
        X0s[(mt * 5 + (n >> 5)) * 512 + swzA(mb + r, n & 31)] =
            (_Float16)fmaxf(acc[t][r] + bias, 0.f);
    }
  }
  // concat wts into K-block 4 (cols 128..158, col 159 zero)
  for (int i = tid; i < 64 * 32; i += 256) {
    int r = i >> 5, j = i & 31;
    float v = (j < 31) ? state[(size_t)(r0 + r) * 1531 + 1500 + j] : 0.f;
    X0s[((r >> 4) * 5 + 4) * 512 + swzA(r & 15, j)] = (_Float16)v;
  }
  __syncthreads();

  // ---- init1: (64x159) @ (159x128), K padded to 160 ----
  {
    v8f acc[4] = {};
    for (int kk = 0; kk < 5; ++kk) {
      for (int i = tid; i < 32 * 128; i += 256) {
        int kr = i >> 7, n = i & 127, k = kk * 32 + kr;
        sBs[(n >> 4) * 512 + swzB(kr, n & 15)] =
            (k < 159) ? (_Float16)iw1[(size_t)k * 128 + n] : (_Float16)0.f;
      }
      __syncthreads();
      v16h Am = ldtile(X0s + (mt * 5 + kk) * 512);
#pragma unroll
      for (int t = 0; t < 4; ++t) {
        int nt = (WAVE >> 2) + 2 * t;
        acc[t] = wmma_f16(Am, ldtile(sBs + nt * 512), acc[t]);
      }
      __syncthreads();
    }
#pragma unroll
    for (int t = 0; t < 4; ++t) {
      int nt = (WAVE >> 2) + 2 * t;
      int n = nt * 16 + (LANE & 15), mb = (LANE & 16) ? 8 : 0;
      float bias = ib1[n];
#pragma unroll
      for (int r = 0; r < 8; ++r)
        X1s[(mt * 4 + (n >> 5)) * 512 + swzA(mb + r, n & 31)] =
            (_Float16)fmaxf(acc[t][r] + bias, 0.f);
    }
  }
  __syncthreads();

  // ---- init2: (64x128) @ (128x64) ----
  {
    v8f acc[2] = {};
    for (int kk = 0; kk < 4; ++kk) {
      for (int i = tid; i < 32 * 64; i += 256) {
        int kr = i >> 6, n = i & 63;
        sBs[(n >> 4) * 512 + swzB(kr, n & 15)] =
            (_Float16)iw2[(size_t)(kk * 32 + kr) * 64 + n];
      }
      __syncthreads();
      v16h Am = ldtile(X1s + (mt * 4 + kk) * 512);
#pragma unroll
      for (int t = 0; t < 2; ++t) {
        int nt = (WAVE >> 2) + 2 * t;
        acc[t] = wmma_f16(Am, ldtile(sBs + nt * 512), acc[t]);
      }
      __syncthreads();
    }
#pragma unroll
    for (int t = 0; t < 2; ++t) {
      int nt = (WAVE >> 2) + 2 * t;
      int n = nt * 16 + (LANE & 15), mb = (LANE & 16) ? 8 : 0;
      float bias = ib2[n];
#pragma unroll
      for (int r = 0; r < 8; ++r)
        X0s[(mt * 5 + (n >> 5)) * 512 + swzA(mb + r, n & 31)] =
            (_Float16)fmaxf(acc[t][r] + bias, 0.f);
    }
  }
  __syncthreads();

  // ---- init3: (64x64) @ (64x31) -> a0 (N padded to 32) ----
  {
    v8f acc = {};
    const int nt = WAVE >> 2;
    for (int kk = 0; kk < 2; ++kk) {
      for (int i = tid; i < 32 * 32; i += 256) {
        int kr = i >> 5, n = i & 31;
        sBs[(n >> 4) * 512 + swzB(kr, n & 15)] =
            (n < 31) ? (_Float16)iw3[(size_t)(kk * 32 + kr) * 31 + n] : (_Float16)0.f;
      }
      __syncthreads();
      acc = wmma_f16(ldtile(X0s + (mt * 5 + kk) * 512), ldtile(sBs + nt * 512), acc);
      __syncthreads();
    }
    int n = nt * 16 + (LANE & 15), mb = mt * 16 + ((LANE & 16) ? 8 : 0);
    float bias = (n < 31) ? ib3[n] : 0.f;
#pragma unroll
    for (int r = 0; r < 8; ++r) {
      int row = mb + r;
      float y = acc[r] + bias;
      a0s[row * 32 + n] = (n < 31) ? y : 0.f;
      if (n < 31) traj[((size_t)(r0 + row) * 31 + 0) * 31 + n] = y;
    }
  }
  __syncthreads();

  // ---- diffusion head (scalar, one row per thread 0..63) ----
  if (tid < 64) {
    int row = tid, gb = r0 + row;
    float aa[31];
#pragma unroll
    for (int i = 0; i < 31; ++i) aa[i] = a0s[row * 32 + i];
    float z[64];
    for (int j = 0; j < 64; ++j) {
      float s = dfb1[j];
      for (int i = 0; i < 31; ++i) s += aa[i] * dfw1[i * 64 + j];
      z[j] = s;
    }
    float m = 0.f;
    for (int j = 0; j < 64; ++j) m += z[j];
    m *= (1.f / 64.f);
    float v = 0.f;
    for (int j = 0; j < 64; ++j) { float d = z[j] - m; v += d * d; }
    float rv = rsqrtf(v * (1.f / 64.f) + 1e-5f);
    for (int j = 0; j < 64; ++j)
      z[j] = fmaxf((z[j] - m) * rv * dflg[j] + dflb[j], 0.f);
    for (int n = 0; n < 31; ++n) {
      float s = dfb2[n];
      for (int j = 0; j < 64; ++j) s += z[j] * dfw2[j * 31 + n];
      sigma_ws[(size_t)gb * 31 + n] = (s > 20.f) ? s : log1pf(expf(s));
    }
  }
}

// ---------------------------------------------------------------------------
// Kernel 3: 30-step SDE scan, drift MLP 31->64->64->31 with two LayerNorms,
// one batch row per thread, weights in LDS. Latency-bound; scalar is right.
// ---------------------------------------------------------------------------
__device__ __forceinline__ void ln_relu64(float* z, const float* g, const float* b) {
  float m = 0.f;
  for (int j = 0; j < 64; ++j) m += z[j];
  m *= (1.f / 64.f);
  float v = 0.f;
  for (int j = 0; j < 64; ++j) { float d = z[j] - m; v += d * d; }
  float rv = rsqrtf(v * (1.f / 64.f) + 1e-5f);
  for (int j = 0; j < 64; ++j) z[j] = fmaxf((z[j] - m) * rv * g[j] + b[j], 0.f);
}

__global__ __launch_bounds__(256) void k_sde(
    const float* __restrict__ noise,
    const float* __restrict__ dw1, const float* __restrict__ db1,
    const float* __restrict__ l1g, const float* __restrict__ l1b,
    const float* __restrict__ dw2, const float* __restrict__ db2,
    const float* __restrict__ l2g, const float* __restrict__ l2b,
    const float* __restrict__ dw3, const float* __restrict__ db3,
    const float* __restrict__ sigma_ws, float* __restrict__ traj) {
  __shared__ float w1[31 * 64], w2[64 * 64], w3[64 * 31];
  __shared__ float b1[64], g1[64], e1[64], b2[64], g2[64], e2[64], b3[31];
  const int tid = threadIdx.x;
  for (int i = tid; i < 31 * 64; i += 256) w1[i] = dw1[i];
  for (int i = tid; i < 64 * 64; i += 256) w2[i] = dw2[i];
  for (int i = tid; i < 64 * 31; i += 256) w3[i] = dw3[i];
  if (tid < 64) {
    b1[tid] = db1[tid]; g1[tid] = l1g[tid]; e1[tid] = l1b[tid];
    b2[tid] = db2[tid]; g2[tid] = l2g[tid]; e2[tid] = l2b[tid];
  }
  if (tid < 31) b3[tid] = db3[tid];
  __syncthreads();

  const int b = blockIdx.x * 256 + tid;
  const float ns = 0.13524003f;  // (1/30)^(1/1.7)
  const float dt = 1.f / 30.f;
  float a[31], sg[31];
#pragma unroll
  for (int i = 0; i < 31; ++i) a[i] = traj[((size_t)b * 31) * 31 + i];
#pragma unroll
  for (int i = 0; i < 31; ++i) sg[i] = sigma_ws[(size_t)b * 31 + i] * ns;

  for (int t = 0; t < 30; ++t) {
    float h1[64];
    for (int j = 0; j < 64; ++j) {
      float s = b1[j];
      for (int i = 0; i < 31; ++i) s += a[i] * w1[i * 64 + j];
      h1[j] = s;
    }
    ln_relu64(h1, g1, e1);
    float h2[64];
    for (int j = 0; j < 64; ++j) {
      float s = b2[j];
      for (int i = 0; i < 64; ++i) s += h1[i] * w2[i * 64 + j];
      h2[j] = s;
    }
    ln_relu64(h2, g2, e2);
    for (int n = 0; n < 31; ++n) {
      float s = b3[n];
      for (int j = 0; j < 64; ++j) s += h2[j] * w3[j * 31 + n];
      a[n] += tanhf(s) * dt + sg[n] * noise[((size_t)t * 2048 + b) * 31 + n];
      traj[((size_t)b * 31 + (t + 1)) * 31 + n] = a[n];
    }
  }
}

// ---------------------------------------------------------------------------
// Kernel 4: attention tail. Only out[:, -1, :] feeds the head, so compute
// q(last), scores vs 31 k's, softmax, weighted v-sum, out proj, proj+softmax.
// ---------------------------------------------------------------------------
__global__ __launch_bounds__(256) void k_attn(
    const float* __restrict__ traj,
    const float* __restrict__ wq, const float* __restrict__ bq,
    const float* __restrict__ wk, const float* __restrict__ bk,
    const float* __restrict__ wv, const float* __restrict__ bv,
    const float* __restrict__ wo, const float* __restrict__ bo,
    const float* __restrict__ pw, const float* __restrict__ pb,
    float* __restrict__ out) {
  __shared__ float sq[961], sk[961], sv[961], so[961], sp[961];
  __shared__ float qb[31], kb[31], vb[31], ob[31], pbb[31];
  const int tid = threadIdx.x;
  for (int i = tid; i < 961; i += 256) {
    sq[i] = wq[i]; sk[i] = wk[i]; sv[i] = wv[i]; so[i] = wo[i]; sp[i] = pw[i];
  }
  if (tid < 31) {
    qb[tid] = bq[tid]; kb[tid] = bk[tid]; vb[tid] = bv[tid];
    ob[tid] = bo[tid]; pbb[tid] = pb[tid];
  }
  __syncthreads();

  const int b = blockIdx.x * 256 + tid;
  const float* tb = traj + (size_t)b * 31 * 31;
  float tl[31];
#pragma unroll
  for (int d = 0; d < 31; ++d) tl[d] = tb[30 * 31 + d];
  float q[31];
  for (int d = 0; d < 31; ++d) {
    float s = qb[d];
    for (int e = 0; e < 31; ++e) s += tl[e] * sq[e * 31 + d];
    q[d] = s;
  }
  const float scale = rsqrtf(31.f);
  float sc[31];
  for (int t = 0; t < 31; ++t) {
    float tr[31];
    for (int e = 0; e < 31; ++e) tr[e] = tb[t * 31 + e];
    float s = 0.f;
    for (int d = 0; d < 31; ++d) {
      float kt = kb[d];
      for (int e = 0; e < 31; ++e) kt += tr[e] * sk[e * 31 + d];
      s += q[d] * kt;
    }
    sc[t] = s * scale;
  }
  float mx = sc[0];
  for (int t = 1; t < 31; ++t) mx = fmaxf(mx, sc[t]);
  float sm = 0.f;
  for (int t = 0; t < 31; ++t) { sc[t] = expf(sc[t] - mx); sm += sc[t]; }
  float inv = 1.f / sm;
  for (int t = 0; t < 31; ++t) sc[t] *= inv;

  float o1[31];
  for (int d = 0; d < 31; ++d) o1[d] = 0.f;
  for (int t = 0; t < 31; ++t) {
    float tr[31];
    for (int e = 0; e < 31; ++e) tr[e] = tb[t * 31 + e];
    for (int d = 0; d < 31; ++d) {
      float v = vb[d];
      for (int e = 0; e < 31; ++e) v += tr[e] * sv[e * 31 + d];
      o1[d] += sc[t] * v;
    }
  }
  float o2[31];
  for (int d = 0; d < 31; ++d) {
    float s = ob[d];
    for (int e = 0; e < 31; ++e) s += o1[e] * so[e * 31 + d];
    o2[d] = s;
  }
  float sel[31];
  for (int n = 0; n < 31; ++n) {
    float s = pbb[n];
    for (int e = 0; e < 31; ++e) s += o2[e] * sp[e * 31 + n];
    sel[n] = s;
  }
  mx = sel[0];
  for (int n = 1; n < 31; ++n) mx = fmaxf(mx, sel[n]);
  sm = 0.f;
  for (int n = 0; n < 31; ++n) { sel[n] = expf(sel[n] - mx); sm += sel[n]; }
  inv = 1.f / sm;
  for (int n = 0; n < 31; ++n) out[(size_t)b * 31 + n] = sel[n] * inv;
}

// ---------------------------------------------------------------------------
extern "C" void kernel_launch(void* const* d_in, const int* in_sizes, int n_in,
                              void* d_out, int out_size, void* d_ws, size_t ws_size,
                              hipStream_t stream) {
  const float* state = (const float*)d_in[0];
  const float* noise = (const float*)d_in[1];
  const float* c1w = (const float*)d_in[2];  const float* c1b = (const float*)d_in[3];
  const float* b1g = (const float*)d_in[4];  const float* b1b = (const float*)d_in[5];
  const float* c2w = (const float*)d_in[6];  const float* c2b = (const float*)d_in[7];
  const float* b2g = (const float*)d_in[8];  const float* b2b = (const float*)d_in[9];
  const float* c3w = (const float*)d_in[10]; const float* c3b = (const float*)d_in[11];
  const float* b3g = (const float*)d_in[12]; const float* b3b = (const float*)d_in[13];
  const float* fcw = (const float*)d_in[14]; const float* fcb = (const float*)d_in[15];
  const float* iw1 = (const float*)d_in[16]; const float* ib1 = (const float*)d_in[17];
  const float* iw2 = (const float*)d_in[18]; const float* ib2 = (const float*)d_in[19];
  const float* iw3 = (const float*)d_in[20]; const float* ib3 = (const float*)d_in[21];
  const float* dw1 = (const float*)d_in[22]; const float* db1 = (const float*)d_in[23];
  const float* l1g = (const float*)d_in[24]; const float* l1b = (const float*)d_in[25];
  const float* dw2 = (const float*)d_in[26]; const float* db2 = (const float*)d_in[27];
  const float* l2g = (const float*)d_in[28]; const float* l2b = (const float*)d_in[29];
  const float* dw3 = (const float*)d_in[30]; const float* db3 = (const float*)d_in[31];
  const float* dfw1 = (const float*)d_in[32]; const float* dfb1 = (const float*)d_in[33];
  const float* dflg = (const float*)d_in[34]; const float* dflb = (const float*)d_in[35];
  const float* dfw2 = (const float*)d_in[36]; const float* dfb2 = (const float*)d_in[37];
  const float* awq = (const float*)d_in[38]; const float* abq = (const float*)d_in[39];
  const float* awk = (const float*)d_in[40]; const float* abk = (const float*)d_in[41];
  const float* awv = (const float*)d_in[42]; const float* abv = (const float*)d_in[43];
  const float* awo = (const float*)d_in[44]; const float* abo = (const float*)d_in[45];
  const float* apw = (const float*)d_in[46]; const float* apb = (const float*)d_in[47];
  (void)in_sizes; (void)n_in; (void)out_size; (void)ws_size;

  char* ws = (char*)d_ws;
  _Float16* Xf   = (_Float16*)(ws + 0);                  // 2048*1920 f16
  float*    traj = (float*)(ws + (8u << 20));            // 2048*31*31 f32
  float*    sig  = (float*)(ws + (16u << 20) + 262144);  // 2048*31 f32

  k_conv<<<2048, 256, 0, stream>>>(state, c1w, c1b, b1g, b1b, c2w, c2b, b2g, b2b,
                                   c3w, c3b, b3g, b3b, Xf);
  k_head<<<32, 256, 0, stream>>>(Xf, state, fcw, fcb, iw1, ib1, iw2, ib2, iw3, ib3,
                                 dfw1, dfb1, dflg, dflb, dfw2, dfb2, traj, sig);
  k_sde<<<8, 256, 0, stream>>>(noise, dw1, db1, l1g, l1b, dw2, db2, l2g, l2b,
                               dw3, db3, sig, traj);
  k_attn<<<8, 256, 0, stream>>>(traj, awq, abq, awk, abk, awv, abv, awo, abo,
                                apw, apb, (float*)d_out);
}
